// QLinear_29317446762577
// MI455X (gfx1250) — 8-bit matrix operand order repaired and checked by reading, compiled, not run on hardware
//
#include <hip/hip_runtime.h>

typedef __attribute__((ext_vector_type(8))) int v8i;
typedef __attribute__((ext_vector_type(4))) int v4i;

#define M_DIM 8192
#define N_DIM 4096
#define K_DIM 4096

__global__ __launch_bounds__(256)
void quant_rows_kernel(const float* __restrict__ src,
                       signed char* __restrict__ q,
                       float* __restrict__ scale) {
    __shared__ float red[256];
    const int row = blockIdx.x;
    const int t   = threadIdx.x;
    const float* r = src + (size_t)row * K_DIM;

    float4 f[4];
    const float4* r4 = (const float4*)r;
    float m = 0.0f;
#pragma unroll
    for (int i = 0; i < 4; ++i) {
        f[i] = r4[t * 4 + i];
        m = fmaxf(m, fmaxf(fmaxf(fabsf(f[i].x), fabsf(f[i].y)),
                           fmaxf(fabsf(f[i].z), fabsf(f[i].w))));
    }
    red[t] = m;
    __syncthreads();
#pragma unroll
    for (int s = 128; s > 0; s >>= 1) {
        if (t < s) red[t] = fmaxf(red[t], red[t + s]);
        __syncthreads();
    }
    const float mx  = red[0];
    const float sc  = mx * (1.0f / 127.0f);
    const float inv = (mx > 0.0f) ? (127.0f / mx) : 0.0f;
    if (t == 0) scale[row] = sc;

    union { signed char c[16]; v4i v; } pk;
#pragma unroll
    for (int i = 0; i < 4; ++i) {
        pk.c[4 * i + 0] = (signed char)__float2int_rn(f[i].x * inv);
        pk.c[4 * i + 1] = (signed char)__float2int_rn(f[i].y * inv);
        pk.c[4 * i + 2] = (signed char)__float2int_rn(f[i].z * inv);
        pk.c[4 * i + 3] = (signed char)__float2int_rn(f[i].w * inv);
    }
    ((v4i*)(q + (size_t)row * K_DIM))[t] = pk.v;
}

__global__ __launch_bounds__(256)
void gemm_iu8_kernel(const signed char* __restrict__ xq,
                     const signed char* __restrict__ wq,
                     const float* __restrict__ ascale,
                     const float* __restrict__ wscale,
                     float* __restrict__ out) {
    const int lane  = threadIdx.x & 31;
    const int wave  = threadIdx.x >> 5;
    const int waveM = wave >> 2;
    const int waveN = wave & 3;
    const int lhalf = lane >> 4;
    const int lmod  = lane & 15;

    const int m0 = blockIdx.x * 128 + waveM * 64;
    const int n0 = blockIdx.y * 256 + waveN * 64;

    v8i acc[4][4] = {};

    const signed char* aBase = xq + (size_t)(m0 + lmod) * K_DIM + lhalf * 8;
    const signed char* bBase = wq + (size_t)(n0 + lmod) * K_DIM + lhalf * 8;

    for (int k0 = 0; k0 < K_DIM; k0 += 64) {
        union { v8i v; unsigned long long u[4]; } a[4];
#pragma unroll
        for (int mi = 0; mi < 4; ++mi) {
            const signed char* p = aBase + (size_t)(mi * 16) * K_DIM + k0;
            a[mi].u[0] = *(const unsigned long long*)(p);
            a[mi].u[1] = *(const unsigned long long*)(p + 16);
            a[mi].u[2] = *(const unsigned long long*)(p + 32);
            a[mi].u[3] = *(const unsigned long long*)(p + 48);
        }
        union { v8i v; unsigned long long u[4]; } b[4];
#pragma unroll
        for (int ni = 0; ni < 4; ++ni) {
            const signed char* p = bBase + (size_t)(ni * 16) * K_DIM + k0;
            b[ni].u[0] = *(const unsigned long long*)(p);
            b[ni].u[1] = *(const unsigned long long*)(p + 16);
            b[ni].u[2] = *(const unsigned long long*)(p + 32);
            b[ni].u[3] = *(const unsigned long long*)(p + 48);
        }
#pragma unroll
        for (int mi = 0; mi < 4; ++mi)
#pragma unroll
            for (int ni = 0; ni < 4; ++ni)
                acc[mi][ni] = __builtin_amdgcn_wmma_i32_16x16x64_iu8(
                     true, a[mi].v,
                     true, b[ni].v,
                    acc[mi][ni],
                     false,  false);
    }

    float asc[4][8];
#pragma unroll
    for (int mi = 0; mi < 4; ++mi)
#pragma unroll
        for (int v = 0; v < 8; ++v)
            asc[mi][v] = ascale[m0 + mi * 16 + v + 8 * lhalf];

#pragma unroll
    for (int ni = 0; ni < 4; ++ni) {
        const int col = n0 + ni * 16 + lmod;
        const float ws = wscale[col];
#pragma unroll
        for (int mi = 0; mi < 4; ++mi) {
#pragma unroll
            for (int v = 0; v < 8; ++v) {
                const int row = m0 + mi * 16 + v + 8 * lhalf;
                const float val = (float)acc[mi][ni][v] * asc[mi][v] * ws;
                out[(size_t)row * N_DIM + col] = (float)(_Float16)val;
            }
        }
    }
}

extern "C" void kernel_launch(void* const* d_in, const int* in_sizes, int n_in,
                              void* d_out, int out_size, void* d_ws, size_t ws_size,
                              hipStream_t stream) {
    const float* input_act = (const float*)d_in[0];
    const float* weight    = (const float*)d_in[1];
    float* out             = (float*)d_out;

    char* ws = (char*)d_ws;
    const size_t XQ_OFF = 0;
    const size_t WQ_OFF = (size_t)M_DIM * K_DIM;
    const size_t AS_OFF = WQ_OFF + (size_t)N_DIM * K_DIM;
    const size_t WS_OFF = AS_OFF + (size_t)M_DIM * sizeof(float);

    signed char* xq = (signed char*)(ws + XQ_OFF);
    signed char* wq = (signed char*)(ws + WQ_OFF);
    float* asc      = (float*)(ws + AS_OFF);
    float* wsc      = (float*)(ws + WS_OFF);

    quant_rows_kernel<<<M_DIM, 256, 0, stream>>>(input_act, xq, asc);
    quant_rows_kernel<<<N_DIM, 256, 0, stream>>>(weight, wq, wsc);
    dim3 grid(M_DIM / 128, N_DIM / 256);
    gemm_iu8_kernel<<<grid, 256, 0, stream>>>(xq, wq, asc, wsc, out);
}
